// ScaledDotProductAttention_41669772706262
// MI455X (gfx1250) — matrix operand order repaired and checked by reading, not compiled, not run on hardware
//
#include <hip/hip_runtime.h>

typedef __attribute__((ext_vector_type(16))) _Float16 v16h;
typedef __attribute__((ext_vector_type(8)))  _Float16 v8h;
typedef __attribute__((ext_vector_type(8)))  float    v8f;
typedef __attribute__((ext_vector_type(4)))  float    v4f;

#define NB 16
#define LQ 2048
#define LK 2048
#define DH 64
#define NT (LK / 32)

// 1/(TEMPERATURE + EPS)
#define INV_T (1.0f / (8.0f + 1e-5f))

// ---------------------------------------------------------------------------
// Pre-pass: K -> f16 row-major (32-key tile = contiguous 4KB);
//           V -> f16 transpose-blocked [b][kt][d][32] (tile = contiguous 4KB)
// 262144 threads, 8 halfs each.
// ---------------------------------------------------------------------------
__global__ __launch_bounds__(256)
void convert_kv_kernel(const float* __restrict__ k, const float* __restrict__ v,
                       _Float16* __restrict__ kh, _Float16* __restrict__ vtb)
{
    const int u = blockIdx.x * blockDim.x + threadIdx.x;   // 0..262143

    // K: 8 consecutive elements (coalesced both sides)
    {
        const size_t base = (size_t)u * 8;
        v4f a0 = *(const v4f*)(k + base);
        v4f a1 = *(const v4f*)(k + base + 4);
        v8h h;
        #pragma unroll
        for (int i = 0; i < 4; ++i) { h[i] = (_Float16)a0[i]; h[4 + i] = (_Float16)a1[i]; }
        *(v8h*)(kh + base) = h;
    }

    // V: output-ordered (coalesced stores; strided reads served by L2)
    {
        const int kkg = (u & 3) * 8;        // key group of 8 within tile
        const int d   = (u >> 2) & 63;
        const int kt  = (u >> 8) & 63;
        const int b   = u >> 14;
        v8h h;
        #pragma unroll
        for (int i = 0; i < 8; ++i) {
            h[i] = (_Float16)v[((size_t)b * LK + (size_t)(kt * 32 + kkg + i)) * DH + d];
        }
        *(v8h*)(vtb + ((((size_t)b * NT + kt) * DH + d) * 32 + kkg)) = h;
    }
}

// ---------------------------------------------------------------------------
// Main attention kernel: async f16 tile copies to LDS, double-buffered.
// ---------------------------------------------------------------------------
__global__ __launch_bounds__(256)
void attn_async_kernel(const float* __restrict__ q,
                       const int*   __restrict__ mask,
                       const _Float16* __restrict__ kh,   // [B][LK][DH]
                       const _Float16* __restrict__ vtb,  // [B][NT][DH][32]
                       float*       __restrict__ out)
{
    __shared__ _Float16 sK [2][32][64];   // f16 K tile, double buffered (2x4KB)
    __shared__ _Float16 sVt[2][64][32];   // f16 V^T tile, double buffered (2x4KB)
    __shared__ _Float16 sP [8][16][32];   // per-wave P tile [q][key] (8KB)

    const int tid  = threadIdx.x;
    const int wave = tid >> 5;
    const int lane = tid & 31;
    const int n    = lane & 15;        // tile column index (N)
    const bool lo  = lane < 16;        // lane half selects K-range / M-range

    const int b    = blockIdx.x >> 4;
    const int qblk = blockIdx.x & 15;
    const int q0   = qblk * 128 + wave * 16;

    const float*    Qb = q    + (size_t)b * LQ * DH;
    const int*      Mb = mask + (size_t)b * LQ * LK;
    float*          Ob = out  + (size_t)b * LQ * DH;
    const _Float16* Kh = kh   + (size_t)b * LK * DH;          // tile kt at +kt*2048 halfs
    const _Float16* Vt = vtb  + (size_t)b * NT * DH * 32;     // tile kt at +kt*2048 halfs

    // LDS byte offsets (flat LDS aperture: low 32 bits = LDS offset)
    const unsigned sK_off  = (unsigned)(size_t)&sK[0][0][0]  + (unsigned)tid * 16u;
    const unsigned sVt_off = (unsigned)(size_t)&sVt[0][0][0] + (unsigned)tid * 16u;
    const unsigned voff    = (unsigned)tid * 16u;             // byte slice within 4KB tile

    // ---- load this wave's Q tile (16x64) into two f16 A-fragments ----
    v16h aq[2];
    {
        const float* qr = Qb + (size_t)(q0 + n) * DH;
        #pragma unroll
        for (int c = 0; c < 2; ++c) {
            const int d0 = c * 32 + (lo ? 0 : 16);
            const int d1 = c * 32 + (lo ? 8 : 24);
            v4f f0 = *(const v4f*)(qr + d0);
            v4f f1 = *(const v4f*)(qr + d0 + 4);
            v4f f2 = *(const v4f*)(qr + d1);
            v4f f3 = *(const v4f*)(qr + d1 + 4);
            v16h a;
            #pragma unroll
            for (int i = 0; i < 4; ++i) {
                a[i]      = (_Float16)f0[i];
                a[4 + i]  = (_Float16)f1[i];
                a[8 + i]  = (_Float16)f2[i];
                a[12 + i] = (_Float16)f3[i];
            }
            aq[c] = a;
        }
    }

    v8f  acc[4] = {};
    float dn = 0.0f;

    // async-copy helper: one b128 per thread per tile (8 waves cover 4KB)
    #define ISSUE_TILE(kt_, buf_)                                               \
        do {                                                                    \
            unsigned long long kg = (unsigned long long)(size_t)                \
                                    (Kh + (size_t)(kt_) * 32 * DH);             \
            unsigned long long vg = (unsigned long long)(size_t)                \
                                    (Vt + (size_t)(kt_) * DH * 32);             \
            asm volatile("global_load_async_to_lds_b128 %0, %1, %2"             \
                         :: "v"(sK_off + (unsigned)(buf_) * 4096u),             \
                            "v"(voff), "s"(kg) : "memory");                     \
            asm volatile("global_load_async_to_lds_b128 %0, %1, %2"             \
                         :: "v"(sVt_off + (unsigned)(buf_) * 4096u),            \
                            "v"(voff), "s"(vg) : "memory");                     \
        } while (0)

    ISSUE_TILE(0, 0);   // prologue

    for (int kt = 0; kt < NT; ++kt) {
        const int cur = kt & 1;
        const int k0  = kt * 32;

        if (kt + 1 < NT) {
            ISSUE_TILE(kt + 1, cur ^ 1);   // overlap next copy with this compute
            asm volatile("s_wait_asynccnt 0x2" ::: "memory");
        } else {
            asm volatile("s_wait_asynccnt 0x0" ::: "memory");
        }
        __syncthreads();   // every thread's slice of the current tile is in LDS

        // ---- S = Q K^T for two 16-key groups; masked clamped exp -> sP ----
        #pragma unroll
        for (int g = 0; g < 2; ++g) {
            const int key = g * 16 + n;
            v16h bk0 = *(const v16h*)&sK[cur][key][(lo ? 0 : 16)];
            v16h bk1 = *(const v16h*)&sK[cur][key][32 + (lo ? 0 : 16)];
            v8f cS = {};
            cS = __builtin_amdgcn_wmma_f32_16x16x32_f16(false, aq[0], false, bk0,
                                                        (short)0, cS, false, false);
            cS = __builtin_amdgcn_wmma_f32_16x16x32_f16(false, aq[1], false, bk1,
                                                        (short)0, cS, false, false);
            const int* mrow = Mb + (size_t)(q0 + (lo ? 0 : 8)) * LK + (k0 + key);
            #pragma unroll
            for (int j = 0; j < 8; ++j) {
                float s = cS[j] * INV_T;
                float e = __expf(s);
                e = fminf(fmaxf(e, 1e-5f), 1e5f);
                int mkv = __builtin_nontemporal_load(mrow + (size_t)j * LK);
                float p = e * (float)mkv;
                sP[wave][j + (lo ? 0 : 8)][key] = (_Float16)p;
            }
        }

        // ---- re-read P in A-layout (same-wave LDS RAW is in-order) ----
        v8h p0 = *(const v8h*)&sP[wave][n][(lo ? 0 : 16)];
        v8h p1 = *(const v8h*)&sP[wave][n][(lo ? 8 : 24)];
        v16h ap = __builtin_shufflevector(p0, p1, 0, 1, 2, 3, 4, 5, 6, 7,
                                          8, 9, 10, 11, 12, 13, 14, 15);
        #pragma unroll
        for (int i = 0; i < 8; ++i) { dn += (float)p0[i] + (float)p1[i]; }

        // ---- acc += P . V ----
        #pragma unroll
        for (int c = 0; c < 4; ++c) {
            v16h bv = *(const v16h*)&sVt[cur][c * 16 + n][(lo ? 0 : 16)];
            acc[c] = __builtin_amdgcn_wmma_f32_16x16x32_f16(false, ap, false, bv,
                                                            (short)0, acc[c], false, false);
        }
        __syncthreads();   // done reading buf cur -> safe to overwrite next iter
    }
    #undef ISSUE_TILE

    // ---- finalize ----
    {
        int di = __builtin_bit_cast(int, dn);
        int dx = __builtin_amdgcn_ds_swizzle(di, 0x401F);  // SWAPX16
        dn += __builtin_bit_cast(float, dx);
    }
    #pragma unroll
    for (int j = 0; j < 8; ++j) {
        const int row = j + (lo ? 0 : 8);
        int dt = __builtin_amdgcn_ds_bpermute(row << 2, __builtin_bit_cast(int, dn));
        float rinv = 1.0f / (__builtin_bit_cast(float, dt) + 1e-5f);
        float* orow = Ob + (size_t)(q0 + row) * DH + n;
        #pragma unroll
        for (int c = 0; c < 4; ++c) orow[c * 16] = acc[c][j] * rinv;
    }
}

// ---------------------------------------------------------------------------
// Fallback (no workspace): self-staging kernel from the previous round.
// ---------------------------------------------------------------------------
__global__ __launch_bounds__(256)
void attn_fallback_kernel(const float* __restrict__ q,
                          const float* __restrict__ k,
                          const float* __restrict__ v,
                          const int*   __restrict__ mask,
                          float*       __restrict__ out)
{
    __shared__ _Float16 sK [32][64];
    __shared__ _Float16 sVt[64][32];
    __shared__ _Float16 sP [8][16][32];

    const int tid  = threadIdx.x;
    const int wave = tid >> 5;
    const int lane = tid & 31;
    const int n    = lane & 15;
    const bool lo  = lane < 16;

    const int b    = blockIdx.x >> 4;
    const int qblk = blockIdx.x & 15;
    const int q0   = qblk * 128 + wave * 16;

    const float* Qb = q    + (size_t)b * LQ * DH;
    const float* Kb = k    + (size_t)b * LK * DH;
    const float* Vb = v    + (size_t)b * LK * DH;
    const int*   Mb = mask + (size_t)b * LQ * LK;
    float*       Ob = out  + (size_t)b * LQ * DH;

    v16h aq[2];
    {
        const float* qr = Qb + (size_t)(q0 + n) * DH;
        #pragma unroll
        for (int c = 0; c < 2; ++c) {
            const int d0 = c * 32 + (lo ? 0 : 16);
            const int d1 = c * 32 + (lo ? 8 : 24);
            v4f f0 = *(const v4f*)(qr + d0);
            v4f f1 = *(const v4f*)(qr + d0 + 4);
            v4f f2 = *(const v4f*)(qr + d1);
            v4f f3 = *(const v4f*)(qr + d1 + 4);
            v16h a;
            #pragma unroll
            for (int i = 0; i < 4; ++i) {
                a[i] = (_Float16)f0[i]; a[4 + i] = (_Float16)f1[i];
                a[8 + i] = (_Float16)f2[i]; a[12 + i] = (_Float16)f3[i];
            }
            aq[c] = a;
        }
    }

    v8f  acc[4] = {};
    float dn = 0.0f;
    const int rr = tid >> 3;
    const int cc = (tid & 7) * 8;

    for (int kt = 0; kt < NT; ++kt) {
        const int k0 = kt * 32;
        {
            const float* kr = Kb + (size_t)(k0 + rr) * DH + cc;
            v4f a0 = *(const v4f*)(kr);
            v4f a1 = *(const v4f*)(kr + 4);
            v8h h;
            #pragma unroll
            for (int i = 0; i < 4; ++i) { h[i] = (_Float16)a0[i]; h[4 + i] = (_Float16)a1[i]; }
            *(v8h*)&sK[rr][cc] = h;

            const float* vr = Vb + (size_t)(k0 + rr) * DH + cc;
            v4f b0 = *(const v4f*)(vr);
            v4f b1 = *(const v4f*)(vr + 4);
            #pragma unroll
            for (int i = 0; i < 4; ++i) {
                sVt[cc + i][rr]     = (_Float16)b0[i];
                sVt[cc + 4 + i][rr] = (_Float16)b1[i];
            }
        }
        __syncthreads();

        #pragma unroll
        for (int g = 0; g < 2; ++g) {
            const int key = g * 16 + n;
            v16h bk0 = *(const v16h*)&sK[key][(lo ? 0 : 16)];
            v16h bk1 = *(const v16h*)&sK[key][32 + (lo ? 0 : 16)];
            v8f cS = {};
            cS = __builtin_amdgcn_wmma_f32_16x16x32_f16(false, aq[0], false, bk0,
                                                        (short)0, cS, false, false);
            cS = __builtin_amdgcn_wmma_f32_16x16x32_f16(false, aq[1], false, bk1,
                                                        (short)0, cS, false, false);
            const int* mrow = Mb + (size_t)(q0 + (lo ? 0 : 8)) * LK + (k0 + key);
            #pragma unroll
            for (int j = 0; j < 8; ++j) {
                float s = cS[j] * INV_T;
                float e = __expf(s);
                e = fminf(fmaxf(e, 1e-5f), 1e5f);
                int mkv = __builtin_nontemporal_load(mrow + (size_t)j * LK);
                sP[wave][j + (lo ? 0 : 8)][key] = (_Float16)(e * (float)mkv);
            }
        }

        v8h p0 = *(const v8h*)&sP[wave][n][(lo ? 0 : 16)];
        v8h p1 = *(const v8h*)&sP[wave][n][(lo ? 8 : 24)];
        v16h ap = __builtin_shufflevector(p0, p1, 0, 1, 2, 3, 4, 5, 6, 7,
                                          8, 9, 10, 11, 12, 13, 14, 15);
        #pragma unroll
        for (int i = 0; i < 8; ++i) { dn += (float)p0[i] + (float)p1[i]; }

        #pragma unroll
        for (int c = 0; c < 4; ++c) {
            v16h bv = *(const v16h*)&sVt[c * 16 + n][(lo ? 0 : 16)];
            acc[c] = __builtin_amdgcn_wmma_f32_16x16x32_f16(false, ap, false, bv,
                                                            (short)0, acc[c], false, false);
        }
        __syncthreads();
    }

    {
        int di = __builtin_bit_cast(int, dn);
        int dx = __builtin_amdgcn_ds_swizzle(di, 0x401F);
        dn += __builtin_bit_cast(float, dx);
    }
    #pragma unroll
    for (int j = 0; j < 8; ++j) {
        const int row = j + (lo ? 0 : 8);
        int dt = __builtin_amdgcn_ds_bpermute(row << 2, __builtin_bit_cast(int, dn));
        float rinv = 1.0f / (__builtin_bit_cast(float, dt) + 1e-5f);
        float* orow = Ob + (size_t)(q0 + row) * DH + n;
        #pragma unroll
        for (int c = 0; c < 4; ++c) orow[c * 16] = acc[c][j] * rinv;
    }
}

extern "C" void kernel_launch(void* const* d_in, const int* in_sizes, int n_in,
                              void* d_out, int out_size, void* d_ws, size_t ws_size,
                              hipStream_t stream) {
    (void)in_sizes; (void)n_in; (void)out_size;
    const float* q    = (const float*)d_in[0];
    const float* k    = (const float*)d_in[1];
    const float* v    = (const float*)d_in[2];
    const int*   mask = (const int*)d_in[3];
    float* out = (float*)d_out;

    const size_t kv_halfs = (size_t)NB * LK * DH;            // per converted matrix
    const size_t ws_need  = 2 * kv_halfs * sizeof(_Float16); // 8 MB

    dim3 grid(NB * (LQ / 128));
    dim3 block(256);

    if (ws_size >= ws_need) {
        _Float16* kh  = (_Float16*)d_ws;
        _Float16* vtb = kh + kv_halfs;
        convert_kv_kernel<<<dim3(1024), block, 0, stream>>>(k, v, kh, vtb);
        attn_async_kernel<<<grid, block, 0, stream>>>(q, mask, kh, vtb, out);
    } else {
        attn_fallback_kernel<<<grid, block, 0, stream>>>(q, k, v, mask, out);
    }
}
